// ONOBlock_61220463837315
// MI455X (gfx1250) — matrix operand order repaired and checked by reading, not compiled, not run on hardware
//
#include <hip/hip_runtime.h>
#include <math.h>
#include <stdint.h>

// ---------------------------------------------------------------------------
// ONOBlock for MI455X (gfx1250): linear-attention transformer block + fused
// sigmoid((X X^T) F) quadratic update. All GEMMs on v_wmma_f32_16x16x32_bf16.
// LDS tiles padded to stride 136 (272B = 68 dwords == 4 mod 64 banks) to kill
// the 16-way bank conflicts a 256B stride would cause, while keeping 16B
// alignment for ds_load_b128 fragment loads. k4 uses the Tensor Data Mover
// (tensor_load_to_lds) to DMA its X tiles into LDS with hardware row padding.
// ---------------------------------------------------------------------------

#define B_ 4
#define N_ 4096
#define D_ 128
#define H_ 8
#define DH_ 16
#define RPB 128                 // rows per block
#define NBLK (N_ / RPB)         // 32 row-blocks per batch
#define LD  136                 // padded LDS row stride (elements)

typedef __attribute__((ext_vector_type(16))) __bf16 v16bf;
typedef __attribute__((ext_vector_type(8)))  __bf16 v8bf;
typedef __attribute__((ext_vector_type(8)))  float  v8f;
typedef __attribute__((ext_vector_type(4)))  unsigned int v4u;
typedef __attribute__((ext_vector_type(8)))  int v8i;
typedef __attribute__((ext_vector_type(4)))  int v4i;

#ifndef __has_builtin
#define __has_builtin(x) 0
#endif
#if __has_builtin(__builtin_amdgcn_tensor_load_to_lds) && \
    __has_builtin(__builtin_amdgcn_s_wait_tensorcnt)
#define USE_TDM 1
#else
#define USE_TDM 0
#endif

__device__ __forceinline__ v8f wmma_bf16(v16bf a, v16bf b, v8f c) {
  // 8 args: (neg_a, A, neg_b, B, c_mod, C, reuse_a, reuse_b)
  return __builtin_amdgcn_wmma_f32_16x16x32_bf16(false, a, false, b, (short)0, c,
                                                 false, false);
}

// A fragment (16x32, MxK) from a row-major matrix. Lane L holds row M=L%16;
// low-half lanes: K in [0..7]U[16..23], high-half: K in [8..15]U[24..31].
__device__ __forceinline__ v16bf load_a_rm(const __bf16* src, int ld, int lane) {
  int m  = lane & 15;
  int kb = (lane >> 4) << 3;            // 0 or 8
  const __bf16* p = src + (size_t)m * ld;
  v16bf a;
#pragma unroll
  for (int i = 0; i < 8; ++i) a[i]     = p[kb + i];
#pragma unroll
  for (int i = 0; i < 8; ++i) a[8 + i] = p[kb + 16 + i];
  return a;
}

// A fragment (16x32) where memory is row-major [K][M] (transposed source).
__device__ __forceinline__ v16bf load_a_cm(const __bf16* src, int ld, int lane) {
  int m  = lane & 15;
  int kb = (lane >> 4) << 3;
  v16bf a;
#pragma unroll
  for (int i = 0; i < 8; ++i) a[i]     = src[(size_t)(kb + i) * ld + m];
#pragma unroll
  for (int i = 0; i < 8; ++i) a[8 + i] = src[(size_t)(kb + 16 + i) * ld + m];
  return a;
}

// B fragment (32x16, KxN) whose column n is ROW n of a row-major matrix src
// (i.e. computes @src^T). Lane L: column N=L%16; low-half lanes: K in
// [0..7]U[16..23], high-half: K in [8..15]U[24..31] -> 2x ds_load_b128.
__device__ __forceinline__ v16bf load_b_t(const __bf16* src, int ld, int lane) {
  int n  = lane & 15;
  int kb = (lane >> 4) << 3;            // 0 or 8
  const __bf16* p = src + (size_t)n * ld + kb;
  v16bf b;
#pragma unroll
  for (int i = 0; i < 8; ++i) b[i] = p[i];
#pragma unroll
  for (int i = 0; i < 8; ++i) b[8 + i] = p[16 + i];
  return b;
}

// B fragment (32x16) from a row-major [K][N] matrix (strided; small uses only).
__device__ __forceinline__ v16bf load_b_rm(const __bf16* src, int ld, int lane) {
  int n  = lane & 15;
  int kb = (lane >> 4) << 3;
  v16bf b;
#pragma unroll
  for (int i = 0; i < 8; ++i) b[i] = src[(size_t)(kb + i) * ld + n];
#pragma unroll
  for (int i = 0; i < 8; ++i) b[8 + i] = src[(size_t)(kb + 16 + i) * ld + n];
  return b;
}

// Stage a 128xD_ f32 weight matrix into LDS as bf16 with padded stride LD.
__device__ __forceinline__ void stage_w(__bf16* dst, const float* src, int tid) {
  for (int i = tid; i < D_ * D_ / 4; i += 256) {
    int row = i >> 5, c4 = (i & 31) << 2;           // 32 float4 per row
    const float4 v = *(const float4*)(src + row * D_ + c4);
    __bf16* o = dst + row * LD + c4;
    o[0] = (__bf16)v.x; o[1] = (__bf16)v.y; o[2] = (__bf16)v.z; o[3] = (__bf16)v.w;
  }
}

// ---------------------------------------------------------------------------
// k0: convert fx -> bf16 once per call (deterministic; no stale ws state).
// ---------------------------------------------------------------------------
__global__ void k0_init(const float4* __restrict__ fx4, __bf16* __restrict__ fxh) {
  int i = blockIdx.x * blockDim.x + threadIdx.x;
  if (i < B_ * N_ * D_ / 4) {
    float4 v = fx4[i];
    __bf16* o = fxh + (size_t)i * 4;
    o[0] = (__bf16)v.x; o[1] = (__bf16)v.y; o[2] = (__bf16)v.z; o[3] = (__bf16)v.w;
  }
}

// ---------------------------------------------------------------------------
// k1: LN1 + Q/K/V projections (WMMA) + q-softmax(dh) + exp(k) stats +
//     per-block partial ctx = exp(k)^T v  (WMMA, transposed-A).
// ---------------------------------------------------------------------------
__global__ void k1_ln_qkv(const float* __restrict__ x,
                          const float* __restrict__ ln1_g,
                          const float* __restrict__ ln1_b,
                          const float* __restrict__ wq,
                          const float* __restrict__ wk,
                          const float* __restrict__ wv,
                          __bf16* __restrict__ q_ws,
                          float* __restrict__ ctx_part,
                          float* __restrict__ colsum_part) {
  extern __shared__ char smem[];
  __bf16* wq_s = (__bf16*)smem;                 // 128 x LD each
  __bf16* wk_s = wq_s + D_ * LD;
  __bf16* wv_s = wk_s + D_ * LD;
  __bf16* xn_s = wv_s + D_ * LD;                // LN(x) rows
  __bf16* ke_s = xn_s + RPB * LD;               // exp(k) tile
  __bf16* v_s  = ke_s + RPB * LD;               // v tile
  float*  cw_s = (float*)(v_s + RPB * LD);      // per-wave column sums [8][128]

  const int tid  = threadIdx.x;
  const int lane = tid & 31;
  const int wave = tid >> 5;
  const int b    = blockIdx.x / NBLK;
  const int rblk = blockIdx.x % NBLK;
  const int r0   = rblk * RPB;

  stage_w(wq_s, wq, tid);
  stage_w(wk_s, wk, tid);
  stage_w(wv_s, wv, tid);
  if (tid < RPB) {
    const float* xr = x + ((size_t)b * N_ + r0 + tid) * D_;
    float s = 0.f, ss = 0.f;
    for (int d = 0; d < D_; ++d) { float v = xr[d]; s += v; ss += v * v; }
    float mu  = s * (1.f / D_);
    float var = ss * (1.f / D_) - mu * mu;
    float rs  = rsqrtf(var + 1e-6f);
    for (int d = 0; d < D_; ++d)
      xn_s[tid * LD + d] = (__bf16)((xr[d] - mu) * rs * ln1_g[d] + ln1_b[d]);
  }
  __syncthreads();

  const int rw = wave * 16, col = lane & 15, half = lane >> 4;

  // A fragments of xn are invariant across q/k/v and all 8 heads: hoist.
  v16bf a_xn[4];
#pragma unroll
  for (int kc = 0; kc < 4; ++kc)
    a_xn[kc] = load_a_rm(xn_s + rw * LD + kc * 32, LD, lane);

  // ---- Q = softmax_dh(xn @ wq^T) : each 16-wide tile is one head ----
#pragma unroll 1
  for (int et = 0; et < H_; ++et) {
    v8f acc = {};
#pragma unroll
    for (int kc = 0; kc < 4; ++kc)
      acc = wmma_bf16(a_xn[kc], load_b_t(wq_s + (et * 16) * LD + kc * 32, LD, lane), acc);
#pragma unroll
    for (int r = 0; r < 8; ++r) {
      float vv = acc[r];
      float mx = vv;
#pragma unroll
      for (int m = 1; m < 16; m <<= 1) mx = fmaxf(mx, __shfl_xor(mx, m, 16));
      float e = __expf(vv - mx);
      float sm = e;
#pragma unroll
      for (int m = 1; m < 16; m <<= 1) sm += __shfl_xor(sm, m, 16);
      int row = r0 + rw + r + 8 * half;
      q_ws[((size_t)b * N_ + row) * D_ + et * 16 + col] = (__bf16)(e / sm);
    }
  }

  // ---- exp(K) + per-wave column sums (softmax over n finalized in k2) ----
#pragma unroll 1
  for (int et = 0; et < H_; ++et) {
    v8f acc = {};
#pragma unroll
    for (int kc = 0; kc < 4; ++kc)
      acc = wmma_bf16(a_xn[kc], load_b_t(wk_s + (et * 16) * LD + kc * 32, LD, lane), acc);
    float part = 0.f;
#pragma unroll
    for (int r = 0; r < 8; ++r) {
      float e = __expf(acc[r]);   // logits ~0.2 magnitude: safe without max-sub
      part += e;
      ke_s[(rw + r + 8 * half) * LD + et * 16 + col] = (__bf16)e;
    }
    part += __shfl_xor(part, 16, 32);           // fold the two row-halves
    if (half == 0) cw_s[wave * D_ + et * 16 + col] = part;
  }

  // ---- V ----
#pragma unroll 1
  for (int et = 0; et < H_; ++et) {
    v8f acc = {};
#pragma unroll
    for (int kc = 0; kc < 4; ++kc)
      acc = wmma_bf16(a_xn[kc], load_b_t(wv_s + (et * 16) * LD + kc * 32, LD, lane), acc);
#pragma unroll
    for (int r = 0; r < 8; ++r)
      v_s[(rw + r + 8 * half) * LD + et * 16 + col] = (__bf16)acc[r];
  }
  __syncthreads();

  // ---- partial ctx for head h = wave: exp(k)^T (128-deep) @ v ----
  {
    const int h = wave;
    v8f acc = {};
#pragma unroll
    for (int kc = 0; kc < 4; ++kc) {
      v16bf a  = load_a_cm(ke_s + (kc * 32) * LD + h * 16, LD, lane);
      v16bf bb = load_b_rm(v_s + (kc * 32) * LD + h * 16, LD, lane);
      acc = wmma_bf16(a, bb, acc);
    }
    float* dst = ctx_part + (((size_t)(b * NBLK + rblk) * H_ + h) * (DH_ * DH_));
#pragma unroll
    for (int r = 0; r < 8; ++r)
      dst[(r + 8 * half) * DH_ + col] = acc[r];
  }
  if (tid < D_) {
    float s = 0.f;
    for (int w = 0; w < 8; ++w) s += cw_s[w * D_ + tid];
    colsum_part[(size_t)(b * NBLK + rblk) * D_ + tid] = s;
  }
}

// ---------------------------------------------------------------------------
// k2: deterministic reduction of the 32 partials; normalize ctx -> bf16.
// ---------------------------------------------------------------------------
__global__ void k2_ctx(const float* __restrict__ ctx_part,
                       const float* __restrict__ colsum_part,
                       __bf16* __restrict__ ctxh) {
  int i = blockIdx.x * blockDim.x + threadIdx.x;
  if (i >= B_ * H_ * DH_ * DH_) return;
  int e = i & 15, d = (i >> 4) & 15, h = (i >> 8) & 7, b = i >> 11;
  float s = 0.f, cs = 0.f;
  for (int p = 0; p < NBLK; ++p) {
    s  += ctx_part[(((size_t)(b * NBLK + p) * H_ + h) * 256) + d * 16 + e];
    cs += colsum_part[(size_t)(b * NBLK + p) * D_ + h * 16 + d];
  }
  ctxh[i] = (__bf16)(s / cs);
}

// ---------------------------------------------------------------------------
// k3: attn = q@ctx, out-proj + residual, LN2, exact-GELU MLP + residual.
// ---------------------------------------------------------------------------
__global__ void k3_attn_mlp(const float* __restrict__ x,
                            const __bf16* __restrict__ q_ws,
                            const __bf16* __restrict__ ctxh,
                            const float* __restrict__ wo, const float* __restrict__ bo,
                            const float* __restrict__ ln2_g, const float* __restrict__ ln2_b,
                            const float* __restrict__ w1, const float* __restrict__ b1,
                            const float* __restrict__ w2, const float* __restrict__ b2,
                            float* __restrict__ xout, __bf16* __restrict__ xh) {
  extern __shared__ char smem[];
  __bf16* wo_s = (__bf16*)smem;
  __bf16* w1_s = wo_s + D_ * LD;
  __bf16* w2_s = w1_s + D_ * LD;
  __bf16* at_s = w2_s + D_ * LD;                // attn tile; reused as h tile
  __bf16* y_s  = at_s + RPB * LD;               // LN2 output
  __bf16* cp_s = y_s + RPB * LD;                // ctx padded to K=32: [H][32][16]
  float*  x1_s = (float*)(cp_s + H_ * 32 * 16); // f32 residual tile (stride LD)
  float*  bo_s = x1_s + RPB * LD;
  float*  b1_s = bo_s + D_;
  float*  b2_s = b1_s + D_;
  __bf16* h_s  = at_s;                          // alias: attn dead after P2

  const int tid  = threadIdx.x;
  const int lane = tid & 31;
  const int wave = tid >> 5;
  const int b    = blockIdx.x / NBLK;
  const int r0   = (blockIdx.x % NBLK) * RPB;

  stage_w(wo_s, wo, tid);
  stage_w(w1_s, w1, tid);
  stage_w(w2_s, w2, tid);
  for (int i = tid; i < H_ * 32 * 16; i += 256) {
    int e = i & 15, k = (i >> 4) & 31, h = i >> 9;
    cp_s[i] = (k < 16) ? ctxh[(((size_t)b * H_ + h) * 16 + k) * 16 + e]
                       : (__bf16)0.0f;
  }
  if (tid < D_) { bo_s[tid] = bo[tid]; b1_s[tid] = b1[tid]; b2_s[tid] = b2[tid]; }
  __syncthreads();

  const int rw = wave * 16, col = lane & 15, half = lane >> 4;

  // ---- P1: attn = q @ ctx (K=16 zero-padded to 32, one WMMA per head) ----
#pragma unroll 1
  for (int h = 0; h < H_; ++h) {
    v16bf a;
    {
      int m  = lane & 15;
      int kb = (lane >> 4) << 3;
      const __bf16* p = q_ws + ((size_t)b * N_ + r0 + rw + m) * D_ + h * 16 + kb;
#pragma unroll
      for (int i = 0; i < 8; ++i) a[i] = p[i];             // K in [0..15]
#pragma unroll
      for (int i = 0; i < 8; ++i) a[8 + i] = (__bf16)0.0f; // K >= 16 padded
    }
    v16bf bb = load_b_rm(cp_s + h * 32 * 16, 16, lane);
    v8f acc = {};
    acc = wmma_bf16(a, bb, acc);
#pragma unroll
    for (int r = 0; r < 8; ++r)
      at_s[(rw + r + 8 * half) * LD + h * 16 + col] = (__bf16)acc[r];
  }
  __syncthreads();

  // ---- P2: x1 = x + attn @ wo^T + bo ----
  {
    v16bf a_at[4];
#pragma unroll
    for (int kc = 0; kc < 4; ++kc)
      a_at[kc] = load_a_rm(at_s + rw * LD + kc * 32, LD, lane);
#pragma unroll 1
    for (int et = 0; et < 8; ++et) {
      v8f acc = {};
#pragma unroll
      for (int kc = 0; kc < 4; ++kc)
        acc = wmma_bf16(a_at[kc], load_b_t(wo_s + (et * 16) * LD + kc * 32, LD, lane), acc);
#pragma unroll
      for (int r = 0; r < 8; ++r) {
        int row = rw + r + 8 * half, c = et * 16 + col;
        x1_s[row * LD + c] =
            x[((size_t)b * N_ + r0 + row) * D_ + c] + acc[r] + bo_s[c];
      }
    }
  }
  __syncthreads();

  // ---- P3: LN2 ----
  if (tid < RPB) {
    const float* xr = x1_s + tid * LD;
    float s = 0.f, ss = 0.f;
    for (int d = 0; d < D_; ++d) { float v = xr[d]; s += v; ss += v * v; }
    float mu = s * (1.f / D_), var = ss * (1.f / D_) - mu * mu;
    float rs = rsqrtf(var + 1e-6f);
    for (int d = 0; d < D_; ++d)
      y_s[tid * LD + d] = (__bf16)((xr[d] - mu) * rs * ln2_g[d] + ln2_b[d]);
  }
  __syncthreads();

  // ---- P4: h = gelu_exact(y @ w1^T + b1) ----
  {
    v16bf a_y[4];
#pragma unroll
    for (int kc = 0; kc < 4; ++kc)
      a_y[kc] = load_a_rm(y_s + rw * LD + kc * 32, LD, lane);
#pragma unroll 1
    for (int et = 0; et < 8; ++et) {
      v8f acc = {};
#pragma unroll
      for (int kc = 0; kc < 4; ++kc)
        acc = wmma_bf16(a_y[kc], load_b_t(w1_s + (et * 16) * LD + kc * 32, LD, lane), acc);
#pragma unroll
      for (int r = 0; r < 8; ++r) {
        int row = rw + r + 8 * half, c = et * 16 + col;
        float t = acc[r] + b1_s[c];
        float g = 0.5f * t * (1.f + erff(t * 0.70710678118f));
        h_s[row * LD + c] = (__bf16)g;
      }
    }
  }
  __syncthreads();

  // ---- P5: x2 = x1 + h @ w2^T + b2 -> outputs ----
  {
    v16bf a_h[4];
#pragma unroll
    for (int kc = 0; kc < 4; ++kc)
      a_h[kc] = load_a_rm(h_s + rw * LD + kc * 32, LD, lane);
#pragma unroll 1
    for (int et = 0; et < 8; ++et) {
      v8f acc = {};
#pragma unroll
      for (int kc = 0; kc < 4; ++kc)
        acc = wmma_bf16(a_h[kc], load_b_t(w2_s + (et * 16) * LD + kc * 32, LD, lane), acc);
#pragma unroll
      for (int r = 0; r < 8; ++r) {
        int row = rw + r + 8 * half, c = et * 16 + col;
        float v = x1_s[row * LD + c] + acc[r] + b2_s[c];
        size_t gi = ((size_t)b * N_ + r0 + row) * D_ + c;
        xout[gi] = v;
        xh[gi]   = (__bf16)v;
      }
    }
  }
}

// ---------------------------------------------------------------------------
// k4: fused sigmoid((X X^T) F) — flash-style, gram never materialized.
// X tiles DMA'd into padded LDS by the Tensor Data Mover (pad_interval=64
// dwords, pad_amount=4 dwords -> the 272B pitch the fragment loaders use);
// F tiles staged transposed so B-fragments are contiguous b128 loads.
// ---------------------------------------------------------------------------
__global__ void k4_quad(const __bf16* __restrict__ xh,
                        const __bf16* __restrict__ fxh,
                        float* __restrict__ fxout) {
  extern __shared__ char smem[];
  __bf16* xm_s  = (__bf16*)smem;           // 128 x LD   X column tile
  __bf16* fmT_s = xm_s + RPB * LD;         // 128 x LD   F tile, TRANSPOSED [d][m]
  __bf16* s_s   = fmT_s + RPB * LD;        // 8 waves x 16 x LD  S staging

  const int tid  = threadIdx.x;
  const int lane = tid & 31;
  const int wave = tid >> 5;
  const int b    = blockIdx.x / NBLK;
  const int ri   = (blockIdx.x % NBLK) * RPB + wave * 16;
  const int col  = lane & 15, half = lane >> 4;

  // Xi A-fragments are loop-invariant: keep them in registers.
  v16bf a_x[4];
#pragma unroll
  for (int kc = 0; kc < 4; ++kc)
    a_x[kc] = load_a_rm(xh + ((size_t)b * N_ + ri) * D_ + kc * 32, D_, lane);

  v8f acc[8];
  {
    v8f z = {};
#pragma unroll
    for (int e = 0; e < 8; ++e) acc[e] = z;
  }

  __bf16* sw = s_s + wave * 16 * LD;       // wave-private S region
  const int tc    = tid & 127;             // transpose: column owned by thread
  const int trow0 = (tid >> 7) * 64;       //            64-row strip

#if USE_TDM
  // Tensor DMA descriptor (D#) for one 128x128 bf16 tile -> padded LDS rows.
  // Group0: count=1 | lds_addr | global_addr(57b) | type=2.
  // Group1: data_size=1(2B), pad_enable, pad_interval=5 (64 dwords = 256B row),
  //         pad_amount=3 (4 dwords = 16B) => 272B pitch; dims/tile/stride.
  v4u g0 = {0u, 0u, 0u, 0u};
  v8i g1 = {0, 0, 0, 0, 0, 0, 0, 0};
  v4i g2 = {0, 0, 0, 0};
  v4i g3 = {0, 0, 0, 0};
  g0[0] = 1u;                               // count=1 (valid user descriptor)
  g0[1] = (unsigned)(size_t)(void*)xm_s;    // LDS byte address (addr[31:0])
  g1[0] = (1 << 16) | (1 << 20) | (5 << 22) | (3 << 25);
  g1[1] = (D_ & 0xFFFF) << 16;              // tensor_dim0 = 128   (bits 79:48)
  g1[2] = (N_ & 0xFFFF) << 16;              // tensor_dim1 = 4096  (bits 111:80)
  g1[3] = (RPB & 0xFFFF) << 16;             // tile_dim0 = 128     (bits 127:112)
  g1[4] = RPB;                              // tile_dim1 = 128; tile_dim2 = 0
  g1[5] = D_;                               // tensor_dim0_stride = 128
#endif

  for (int m = 0; m < NBLK; ++m) {
    const __bf16* xm = xh  + ((size_t)b * N_ + m * RPB) * D_;
    const __bf16* fm = fxh + ((size_t)b * N_ + m * RPB) * D_;
#if USE_TDM
    if (wave == 0) {
      uint64_t ga = (uint64_t)(uintptr_t)xm;
      g0[2] = (unsigned)ga;                           // global_addr[31:0]
      g0[3] = (2u << 30) | (unsigned)((ga >> 32) & 0x1FFFFFFu); // [56:32]|type=2
#if __clang_major__ >= 23
      v8i g4 = {0, 0, 0, 0, 0, 0, 0, 0};
      __builtin_amdgcn_tensor_load_to_lds(g0, g1, g2, g3, g4, 0);
#else
      __builtin_amdgcn_tensor_load_to_lds(g0, g1, g2, g3, 0);
#endif
    }
#else
    // Fallback: vectorized copy into padded-stride LDS.
    for (int i = tid; i < RPB * D_ / 8; i += 256) {
      int row = i >> 4, seg = (i & 15) << 3;
      *(v8bf*)(xm_s + row * LD + seg) = *(const v8bf*)(xm + row * D_ + seg);
    }
#endif
    // F tile: transpose into fmT[d][m] so B-fragments are contiguous.
    for (int g = 0; g < 8; ++g) {
      v8bf t;
#pragma unroll
      for (int s = 0; s < 8; ++s)
        t[s] = fm[(size_t)(trow0 + g * 8 + s) * D_ + tc];
      *(v8bf*)(fmT_s + tc * LD + trow0 + g * 8) = t;
    }
    // Prefetch next tile while this one is consumed (global_prefetch_b8).
    if (m + 1 < NBLK) {
      const char* nx = (const char*)(xm + (size_t)RPB * D_) + tid * 128;
      const char* nf = (const char*)(fm + (size_t)RPB * D_) + tid * 128;
      __builtin_prefetch(nx, 0, 1);
      __builtin_prefetch(nf, 0, 1);
    }
#if USE_TDM
    if (wave == 0) __builtin_amdgcn_s_wait_tensorcnt((short)0);
#endif
    __syncthreads();

    // S(16x128) = Xi @ Xm^T  (bf16 stage via wave-private LDS)
#pragma unroll 1
    for (int jt = 0; jt < 8; ++jt) {
      v8f sacc = {};
#pragma unroll
      for (int kc = 0; kc < 4; ++kc)
        sacc = wmma_bf16(a_x[kc], load_b_t(xm_s + (jt * 16) * LD + kc * 32, LD, lane), sacc);
#pragma unroll
      for (int r = 0; r < 8; ++r)
        sw[(r + 8 * half) * LD + jt * 16 + col] = (__bf16)sacc[r];
    }
    // Same-wave DS ops are in-order: no block barrier needed before reading sw.
    v16bf a_s[4];
#pragma unroll
    for (int kc = 0; kc < 4; ++kc)
      a_s[kc] = load_a_rm(sw + kc * 32, LD, lane);

    // acc(16x128) += S @ Fm   (B from transposed Fm: 2x b128 per fragment)
#pragma unroll 1
    for (int et = 0; et < 8; ++et) {
#pragma unroll
      for (int kc = 0; kc < 4; ++kc)
        acc[et] = wmma_bf16(a_s[kc], load_b_t(fmT_s + (et * 16) * LD + kc * 32, LD, lane), acc[et]);
    }
    __syncthreads();
  }

#pragma unroll 1
  for (int et = 0; et < 8; ++et) {
#pragma unroll
    for (int r = 0; r < 8; ++r) {
      int row = ri + r + 8 * half;
      float v = acc[et][r];
      fxout[((size_t)b * N_ + row) * D_ + et * 16 + col] =
          1.f / (1.f + __expf(-v));
    }
  }
}

// ---------------------------------------------------------------------------
extern "C" void kernel_launch(void* const* d_in, const int* in_sizes, int n_in,
                              void* d_out, int out_size, void* d_ws, size_t ws_size,
                              hipStream_t stream) {
  const float* x    = (const float*)d_in[0];
  const float* fx   = (const float*)d_in[1];
  const float* ln1g = (const float*)d_in[2];
  const float* ln1b = (const float*)d_in[3];
  const float* wq   = (const float*)d_in[4];
  const float* wk   = (const float*)d_in[5];
  const float* wv   = (const float*)d_in[6];
  const float* wo   = (const float*)d_in[7];
  const float* bo   = (const float*)d_in[8];
  const float* ln2g = (const float*)d_in[9];
  const float* ln2b = (const float*)d_in[10];
  const float* w1   = (const float*)d_in[11];
  const float* b1   = (const float*)d_in[12];
  const float* w2   = (const float*)d_in[13];
  const float* b2   = (const float*)d_in[14];

  float* xout  = (float*)d_out;
  float* fxout = xout + (size_t)B_ * N_ * D_;

  char* ws = (char*)d_ws;
  __bf16* q_ws = (__bf16*)ws;  ws += sizeof(__bf16) * (size_t)B_ * N_ * D_;
  __bf16* xh   = (__bf16*)ws;  ws += sizeof(__bf16) * (size_t)B_ * N_ * D_;
  __bf16* fxh  = (__bf16*)ws;  ws += sizeof(__bf16) * (size_t)B_ * N_ * D_;
  float* ctx_part    = (float*)ws; ws += sizeof(float) * (size_t)B_ * NBLK * H_ * DH_ * DH_;
  float* colsum_part = (float*)ws; ws += sizeof(float) * (size_t)B_ * NBLK * D_;
  __bf16* ctxh = (__bf16*)ws;  ws += sizeof(__bf16) * (size_t)B_ * H_ * DH_ * DH_;

  const int total4 = B_ * N_ * D_ / 4;
  k0_init<<<(total4 + 255) / 256, 256, 0, stream>>>((const float4*)fx, fxh);

  size_t lds1 = sizeof(__bf16) * (size_t)(6 * RPB * LD) + sizeof(float) * 8 * D_;
  k1_ln_qkv<<<B_ * NBLK, 256, lds1, stream>>>(x, ln1g, ln1b, wq, wk, wv,
                                              q_ws, ctx_part, colsum_part);

  k2_ctx<<<(B_ * H_ * DH_ * DH_ + 255) / 256, 256, 0, stream>>>(ctx_part,
                                                                colsum_part, ctxh);

  size_t lds3 = sizeof(__bf16) * (size_t)(5 * RPB * LD + H_ * 32 * 16) +
                sizeof(float) * (size_t)(RPB * LD + 3 * D_);
  k3_attn_mlp<<<B_ * NBLK, 256, lds3, stream>>>(x, q_ws, ctxh, wo, bo, ln2g, ln2b,
                                                w1, b1, w2, b2, xout, xh);

  size_t lds4 = sizeof(__bf16) * (size_t)(3 * RPB * LD);
  k4_quad<<<B_ * NBLK, 256, lds4, stream>>>(xh, fxh, fxout);
}
